// MultiHeadAttention_50190987821370
// MI455X (gfx1250) — matrix operand order repaired and checked by reading, not compiled, not run on hardware
//
#include <hip/hip_runtime.h>
#include <hip/hip_bf16.h>

#define B_ 2
#define S_ 2048
#define D_ 1024
#define H_ 16
// depth = 64

typedef __attribute__((ext_vector_type(16))) _Float16 v16h;
typedef __attribute__((ext_vector_type(8)))  float    v8f;
typedef __attribute__((ext_vector_type(8)))  _Float16 h8;
typedef __attribute__((ext_vector_type(4)))  _Float16 h4;

#define WMMA(a, b, c) \
  __builtin_amdgcn_wmma_f32_16x16x32_f16(false, (a), false, (b), (short)0, (c), false, false)

// ---- CDNA5 async global->LDS copy (guarded; falls back to load+ds_store) ---
#if defined(__HIP_DEVICE_COMPILE__) && __has_builtin(__builtin_amdgcn_global_load_async_to_lds_b128)
#define HAVE_ASYNC_LDS 1
#else
#define HAVE_ASYNC_LDS 0
#endif

#if HAVE_ASYNC_LDS
typedef int v4i_ __attribute__((vector_size(16)));
typedef __attribute__((address_space(1))) v4i_* as1_v4i_p;
typedef __attribute__((address_space(3))) v4i_* as3_v4i_p;

static __device__ __forceinline__ void async_copy16(const void* g, void* l) {
  // generic->AS1 / AS3 via integer round-trip (low 32 bits of a flat LDS
  // address are the LDS byte offset per the aperture rules)
  __builtin_amdgcn_global_load_async_to_lds_b128(
      (as1_v4i_p)(uintptr_t)g,
      (as3_v4i_p)(uint32_t)(uintptr_t)l,
      0, 0);
}
#endif

// ---- Fragment loaders (contiguous k order, same for A and B) -----------------------
// A 16x32 f16: lane holds row M=laneLo; e -> K = kb + 16*laneHi + e
static __device__ __forceinline__ v16h load_frag_a(const _Float16* __restrict__ base,
                                                   int row, int ld, int kb, int laneHi) {
  const _Float16* p = base + (size_t)row * ld + kb + 16 * laneHi;
  h8 lo = *(const h8*)p;
  h8 hi = *(const h8*)(p + 8);
  v16h a;
#pragma unroll
  for (int i = 0; i < 8; ++i) { a[i] = lo[i]; a[i + 8] = hi[i]; }
  return a;
}

// Same A layout but from LDS with arbitrary (padded) row stride in halves
static __device__ __forceinline__ v16h load_frag_a_lds(const _Float16* base,
                                                       int row, int stride, int kb, int laneHi) {
  const _Float16* p = base + row * stride + kb + 16 * laneHi;
  h8 lo = *(const h8*)p;
  h8 hi = *(const h8*)(p + 8);
  v16h a;
#pragma unroll
  for (int i = 0; i < 8; ++i) { a[i] = lo[i]; a[i + 8] = hi[i]; }
  return a;
}

// B 32x16 f16: lane holds column N=laneLo; e -> K = kb + 16*laneHi + e
static __device__ __forceinline__ v16h load_frag_b(const _Float16* __restrict__ base,
                                                   int row, int ld, int kb, int laneHi) {
  const _Float16* p = base + (size_t)row * ld + kb + 16 * laneHi;
  h8 lo = *(const h8*)p;
  h8 hi = *(const h8*)(p + 8);
  v16h b;
#pragma unroll
  for (int i = 0; i < 8; ++i) { b[i] = lo[i]; b[i + 8] = hi[i]; }
  return b;
}

// ---- Stage 0: f32 -> f16 cast ----------------------------------------------
__global__ __launch_bounds__(256) void cast_f32_f16(const float* __restrict__ in,
                                                    _Float16* __restrict__ out, int n) {
  int i = (blockIdx.x * 256 + threadIdx.x) * 4;
  if (i + 3 < n) {
    float4 v4 = *(const float4*)(in + i);
    h4 o;
    o[0] = (_Float16)v4.x; o[1] = (_Float16)v4.y;
    o[2] = (_Float16)v4.z; o[3] = (_Float16)v4.w;
    *(h4*)(out + i) = o;
  }
}

// ---- Stage 1: W [K,N] f32 -> Wt [N,K] f16 (tiled transpose) ----------------
__global__ __launch_bounds__(256) void transpose_cast(const float* __restrict__ W,
                                                      _Float16* __restrict__ Wt) {
  __shared__ float tile[32][33];
  const int bx = blockIdx.x & (D_ / 32 - 1);   // N tile
  const int by = blockIdx.x >> 5;              // K tile
  const int tx = threadIdx.x & 31;
  const int ty = threadIdx.x >> 5;
#pragma unroll
  for (int i = 0; i < 32; i += 8)
    tile[ty + i][tx] = W[(size_t)(by * 32 + ty + i) * D_ + bx * 32 + tx];
  __syncthreads();
#pragma unroll
  for (int i = 0; i < 32; i += 8)
    Wt[(size_t)(bx * 32 + ty + i) * D_ + by * 32 + tx] = (_Float16)tile[tx][ty + i];
}

// ---- Stage 2/4: GEMM  C[4096,1024] = (X f16 @ W + bias) * oscale -----------
// MODE 0: store f16 head-split  [B,H,S,64]   (Q with oscale, K)
// MODE 1: store f16 transposed  [B,H,64,S]   (V)
// MODE 2: store f32 row-major   [B*S, D]     (final output)
template <int MODE>
__global__ __launch_bounds__(256) void gemm_wmma(const _Float16* __restrict__ X,
                                                 const _Float16* __restrict__ Wt,
                                                 const float* __restrict__ bias,
                                                 void* __restrict__ out, float oscale) {
  const int lane   = threadIdx.x & 31;
  const int laneLo = lane & 15;
  const int laneHi = lane >> 4;
  const int wave   = (blockIdx.x * blockDim.x + threadIdx.x) >> 5;
  const int tileN4 = wave & (D_ / 64 - 1);
  const int tileM  = wave >> 4;
  const int rowA   = tileM * 16 + laneLo;
  const int colB   = tileN4 * 64 + laneLo;

  v8f acc[4] = {};
  v16h a = load_frag_a(X, rowA, D_, 0, laneHi);
  v16h b[4];
#pragma unroll
  for (int nn = 0; nn < 4; ++nn) b[nn] = load_frag_b(Wt, colB + nn * 16, D_, 0, laneHi);

  for (int kb = 0; kb < D_; kb += 32) {
    const int kn = (kb + 32) & (D_ - 1);       // wraps to 0 on last iter (dead)
    v16h a2 = load_frag_a(X, rowA, D_, kn, laneHi);
    v16h b2[4];
#pragma unroll
    for (int nn = 0; nn < 4; ++nn) b2[nn] = load_frag_b(Wt, colB + nn * 16, D_, kn, laneHi);
#pragma unroll
    for (int nn = 0; nn < 4; ++nn) acc[nn] = WMMA(a, b[nn], acc[nn]);
    a = a2;
#pragma unroll
    for (int nn = 0; nn < 4; ++nn) b[nn] = b2[nn];
  }

#pragma unroll
  for (int nn = 0; nn < 4; ++nn) {
    const int col = tileN4 * 64 + nn * 16 + laneLo;
    const float bcol = bias[col];
#pragma unroll
    for (int r = 0; r < 8; ++r) {
      const int row = tileM * 16 + r + 8 * laneHi;
      const float vv = (acc[nn][r] + bcol) * oscale;
      if (MODE == 2) {
        ((float*)out)[(size_t)row * D_ + col] = vv;
      } else {
        const int bb = row >> 11, s = row & (S_ - 1);
        const int h = col >> 6,  d = col & 63;
        if (MODE == 0)
          ((_Float16*)out)[(((size_t)(bb * H_ + h)) * S_ + s) * 64 + d] = (_Float16)vv;
        else
          ((_Float16*)out)[(((size_t)(bb * H_ + h)) * 64 + d) * S_ + s] = (_Float16)vv;
      }
    }
  }
}

// ---- Stage 3: causal flash attention ---------------------------------------
// Block = 8 waves = 8 consecutive q-tiles of ONE (b,h). K (32x64) and V^T
// (64x32) tiles are staged into double-buffered LDS with async global->LDS
// copies: tile k+1 is in flight while tile k is consumed (s_wait_asynccnt 2
// relies on in-order async completion). Transposed-logits trick: S^T = K@Q^T,
// so exp(S^T)'s D-register layout is the B-fragment of P^T for O^T = V^T@P^T
// (one shfl_xor(16) fixes the lane-half K packing difference). The softmax
// scale is pre-folded into Q; masking runs only on diagonal tiles; fully-
// future tiles are skipped wave-uniformly (staging/barriers still executed).
#define KSTRIDE 72   // 32 rows x 64 halves, padded -> conflict-free b128 reads
#define VSTRIDE 40   // 64 rows x 32 halves, padded
__global__ __launch_bounds__(256) void flash_attn_wmma(const _Float16* __restrict__ Qh,
                                                       const _Float16* __restrict__ Kh,
                                                       const _Float16* __restrict__ Vt,
                                                       _Float16* __restrict__ Oh) {
  __shared__ _Float16 Klds[2][32 * KSTRIDE];
  __shared__ _Float16 Vlds[2][64 * VSTRIDE];

  const int tid    = threadIdx.x;
  const int lane   = tid & 31;
  const int laneLo = lane & 15;
  const int laneHi = lane >> 4;
  const int w      = tid >> 5;                  // wave in block: 0..7
  const int bh     = blockIdx.x >> 4;           // 0..31
  const int qgroup = blockIdx.x & 15;           // 0..15
  const int qbase  = (qgroup * 8 + w) * 16;
  const int myq    = qbase + laneLo;
  const int kbMax  = (qgroup * 8 + 7) * 16 + 15;  // uniform per block

  const _Float16* Qb = Qh + (size_t)bh * S_ * 64;
  const _Float16* Kb = Kh + (size_t)bh * S_ * 64;
  const _Float16* Vb = Vt + (size_t)bh * 64 * S_;

  // staging assignment (256 threads -> one 16B chunk of K and one of V each)
  const int kr = tid >> 3, kc = (tid & 7) * 8;   // K: 32 rows x 8 chunks
  const int vr = tid >> 2, vc = (tid & 3) * 8;   // V: 64 rows x 4 chunks

  const v16h qf0 = load_frag_b(Qb, myq, 64, 0,  laneHi);   // Q pre-scaled
  const v16h qf1 = load_frag_b(Qb, myq, 64, 32, laneHi);

  v8f o[4] = {};
  float m = -3.0e38f, lsum = 0.0f;

#if HAVE_ASYNC_LDS
  // prologue: stage tile 0 into buffer 0
  async_copy16(Kb + (size_t)kr * 64 + kc,       &Klds[0][kr * KSTRIDE + kc]);
  async_copy16(Vb + (size_t)vr * S_ + vc,       &Vlds[0][vr * VSTRIDE + vc]);
#endif

  for (int kb = 0; kb <= kbMax; kb += 32) {
    const int cur = (kb >> 5) & 1;
#if HAVE_ASYNC_LDS
    if (kb + 32 <= kbMax) {      // issue next tile into the other buffer
      const int nxt = cur ^ 1;
      async_copy16(Kb + (size_t)(kb + 32 + kr) * 64 + kc, &Klds[nxt][kr * KSTRIDE + kc]);
      async_copy16(Vb + (size_t)vr * S_ + (kb + 32) + vc, &Vlds[nxt][vr * VSTRIDE + vc]);
      __builtin_amdgcn_s_wait_asynccnt(2);   // current tile's 2 copies done
    } else {
      __builtin_amdgcn_s_wait_asynccnt(0);
    }
#else
    {
      h8 tk = *(const h8*)(Kb + (size_t)(kb + kr) * 64 + kc);
      h8 tv = *(const h8*)(Vb + (size_t)vr * S_ + kb + vc);
      *(h8*)&Klds[cur][kr * KSTRIDE + kc] = tk;
      *(h8*)&Vlds[cur][vr * VSTRIDE + vc] = tv;
    }
#endif
    __syncthreads();                           // tile visible to all waves

    if (kb <= qbase + 15) {                    // wave-uniform: skip future tiles
      // preload all fragments (overlapping ds waits)
      v16h ka[4], va[4];
      ka[0] = load_frag_a_lds(Klds[cur], laneLo,      KSTRIDE, 0,  laneHi);
      ka[1] = load_frag_a_lds(Klds[cur], laneLo,      KSTRIDE, 32, laneHi);
      ka[2] = load_frag_a_lds(Klds[cur], 16 + laneLo, KSTRIDE, 0,  laneHi);
      ka[3] = load_frag_a_lds(Klds[cur], 16 + laneLo, KSTRIDE, 32, laneHi);
#pragma unroll
      for (int dt = 0; dt < 4; ++dt)
        va[dt] = load_frag_a_lds(Vlds[cur], dt * 16 + laneLo, VSTRIDE, 0, laneHi);

      // S^T = K_tile @ Q^T (two 16-key subtiles; D layout: lane=q, slots=key)
      v8f st0 = {}, st1 = {};
      st0 = WMMA(ka[0], qf0, st0);
      st0 = WMMA(ka[1], qf1, st0);
      st1 = WMMA(ka[2], qf0, st1);
      st1 = WMMA(ka[3], qf1, st1);

      float s0[8], s1[8];
#pragma unroll
      for (int r = 0; r < 8; ++r) { s0[r] = st0[r]; s1[r] = st1[r]; }
      if (kb + 31 > qbase) {                   // diagonal tile: apply causal mask
#pragma unroll
        for (int r = 0; r < 8; ++r) {
          const int k0 = kb + r + 8 * laneHi;
          if (k0      > myq) s0[r] = -1.0e30f;
          if (k0 + 16 > myq) s1[r] = -1.0e30f;
        }
      }
      float tmax = -3.0e38f;
#pragma unroll
      for (int r = 0; r < 8; ++r) tmax = fmaxf(tmax, fmaxf(s0[r], s1[r]));
      tmax = fmaxf(tmax, __shfl_xor(tmax, 16, 32));
      const float mnew  = fmaxf(m, tmax);
      const float alpha = exp2f(m - mnew);
      float p0[8], p1[8], psum = 0.0f;
#pragma unroll
      for (int r = 0; r < 8; ++r) {
        p0[r] = exp2f(s0[r] - mnew);
        p1[r] = exp2f(s1[r] - mnew);
        psum += p0[r] + p1[r];
      }
      psum += __shfl_xor(psum, 16, 32);
      lsum = lsum * alpha + psum;
      m = mnew;
#pragma unroll
      for (int dt = 0; dt < 4; ++dt)
#pragma unroll
        for (int r = 0; r < 8; ++r) o[dt][r] *= alpha;

      // P^T B-fragment: B packs K as e + 16*laneHi; slots pack r + 8*laneHi
      v16h pf;
#pragma unroll
      for (int e = 0; e < 8; ++e) {
        const float p1sw = __shfl_xor(p1[e], 16, 32);
        const float p0sw = __shfl_xor(p0[e], 16, 32);
        pf[e]     = (_Float16)(laneHi ? p1sw  : p0[e]);
        pf[e + 8] = (_Float16)(laneHi ? p1[e] : p0sw);
      }
      // O^T += V^T(64 x 32 keys) @ P^T
#pragma unroll
      for (int dt = 0; dt < 4; ++dt) o[dt] = WMMA(va[dt], pf, o[dt]);
    }
    __syncthreads();                           // everyone done with buf[cur]
  }

  const float inv = 1.0f / lsum;
  const int b = bh >> 4, h = bh & 15;
  _Float16* orow = Oh + (((size_t)b * S_ + qbase + laneLo) * H_ + h) * 64;
#pragma unroll
  for (int dt = 0; dt < 4; ++dt)
#pragma unroll
    for (int r = 0; r < 8; ++r)
      orow[dt * 16 + r + 8 * laneHi] = (_Float16)(o[dt][r] * inv);
}

// ---- Host launcher ----------------------------------------------------------
extern "C" void kernel_launch(void* const* d_in, const int* in_sizes, int n_in,
                              void* d_out, int out_size, void* d_ws, size_t ws_size,
                              hipStream_t stream) {
  const float* q  = (const float*)d_in[0];
  const float* k  = (const float*)d_in[1];
  const float* v  = (const float*)d_in[2];
  // d_in[3] = causal mask, implemented analytically in the attention kernel
  const float* Wq = (const float*)d_in[4];
  const float* bq = (const float*)d_in[5];
  const float* Wk = (const float*)d_in[6];
  const float* bk = (const float*)d_in[7];
  const float* Wv = (const float*)d_in[8];
  const float* bv = (const float*)d_in[9];
  const float* Wo = (const float*)d_in[10];
  const float* bo = (const float*)d_in[11];

  const size_t NT = (size_t)B_ * S_;       // 4096 tokens
  const size_t WN = (size_t)D_ * D_;
  const size_t XN = NT * D_;

  _Float16* Wqt = (_Float16*)d_ws;
  _Float16* Wkt = Wqt + WN;
  _Float16* Wvt = Wkt + WN;
  _Float16* Wot = Wvt + WN;
  _Float16* qh  = Wot + WN;
  _Float16* kh  = qh + XN;
  _Float16* vh  = kh + XN;
  _Float16* Qh  = vh + XN;                 // head-split [B,H,S,64]
  _Float16* Kh  = Qh + XN;
  _Float16* Vt  = Kh + XN;                 // transposed [B,H,64,S]
  _Float16* Oh  = Vt + XN;                 // attention out [B,S,H*64]

  const int nElem = (int)XN;
  cast_f32_f16<<<nElem / 1024, 256, 0, stream>>>(q, qh, nElem);
  cast_f32_f16<<<nElem / 1024, 256, 0, stream>>>(k, kh, nElem);
  cast_f32_f16<<<nElem / 1024, 256, 0, stream>>>(v, vh, nElem);

  const int tBlocks = (D_ / 32) * (D_ / 32);
  transpose_cast<<<tBlocks, 256, 0, stream>>>(Wq, Wqt);
  transpose_cast<<<tBlocks, 256, 0, stream>>>(Wk, Wkt);
  transpose_cast<<<tBlocks, 256, 0, stream>>>(Wv, Wvt);
  transpose_cast<<<tBlocks, 256, 0, stream>>>(Wo, Wot);

  const float qscale = 0.125f * 1.44269504088896f;   // 1/sqrt(64) * log2(e)
  const int gemmBlocks = (int)((NT / 16) * (D_ / 64) / 8);   // 512
  gemm_wmma<0><<<gemmBlocks, 256, 0, stream>>>(qh, Wqt, bq, (void*)Qh, qscale);
  gemm_wmma<0><<<gemmBlocks, 256, 0, stream>>>(kh, Wkt, bk, (void*)Kh, 1.0f);
  gemm_wmma<1><<<gemmBlocks, 256, 0, stream>>>(vh, Wvt, bv, (void*)Vt, 1.0f);

  const int attnBlocks = B_ * H_ * (S_ / 16) / 8;            // 512
  flash_attn_wmma<<<attnBlocks, 256, 0, stream>>>(Qh, Kh, Vt, Oh);

  gemm_wmma<2><<<gemmBlocks, 256, 0, stream>>>(Oh, Wot, bo, d_out, 1.0f);
}
